// GraphTransformer_44349832298689
// MI455X (gfx1250) — matrix operand order repaired and checked by reading, compiled, not run on hardware
//
#include <hip/hip_runtime.h>
#include <hip/hip_bf16.h>

typedef __attribute__((ext_vector_type(16))) __bf16 v16bf;
typedef __attribute__((ext_vector_type(8)))  __bf16 v8bf;
typedef __attribute__((ext_vector_type(8)))  float  v8f;

#define D_MODEL 128
#define HEADS   8
#define HDIM    16
#define KPAD    136   // LDS row stride (bf16 elems) for transposed W slab

// ---------------------------------------------------------------------------
// Tiled GEMM: out[M,N] = A[M,K] @ W[K,N] (+ bias[N]) (+ pos[N])
// fp32 in/out, bf16 WMMA math with fp32 accumulation.
// Requirements (met by all call sites): N multiple of 128, K multiple of 32.
// Block = 256 threads (8 waves). Block tile = 128 rows x 128 cols.
// Wave w owns rows [w*16, w*16+16) and all 8 column tiles (8 accumulators);
// each A fragment is reused 8x. W is staged TRANSPOSED in LDS as bf16 so each
// lane's B fragment is 16 contiguous elements (2x ds_load_b128).
// ---------------------------------------------------------------------------
__global__ __launch_bounds__(256)
void gemm_wmma_bf16(const float* __restrict__ A, const float* __restrict__ W,
                    const float* __restrict__ bias, const float* __restrict__ pos,
                    float* __restrict__ out, int M, int K, int N) {
    __shared__ __bf16 WldsT[128 * KPAD];  // [col][k] slab, ~34 KB

    const int lane = threadIdx.x & 31;
    const int wave = threadIdx.x >> 5;
    const int half = lane >> 4;   // 16-lane half of the wave
    const int nIn  = lane & 15;   // row (A) / column (B, C/D) within a tile

    const int rowBase = blockIdx.x * 128 + wave * 16;
    const int colBase = blockIdx.y * 128;

    v8f zero = {};
    v8f acc[8];
#pragma unroll
    for (int t = 0; t < 8; ++t) acc[t] = zero;

    // A row pointer for this lane (clamped for M tail; stores are guarded)
    const int ar  = rowBase + nIn;
    const int arc = ar < M ? ar : (M - 1);
    const float* Arow = A + (size_t)arc * K;

    for (int ks = 0; ks < K; ks += 128) {
        const int kslab = (K - ks) < 128 ? (K - ks) : 128;

        // Cooperative transposed staging: WldsT[c][k] = (bf16) W[ks+k][colBase+c]
        // float4 global reads (coalesced), 4 sub-word LDS stores each.
        for (int idx = threadIdx.x; idx < kslab * 32; idx += 256) {
            const int kk  = idx >> 5;
            const int cc4 = (idx & 31) << 2;
            const float4 wv =
                *(const float4*)&W[(size_t)(ks + kk) * N + colBase + cc4];
            WldsT[(cc4 + 0) * KPAD + kk] = (__bf16)wv.x;
            WldsT[(cc4 + 1) * KPAD + kk] = (__bf16)wv.y;
            WldsT[(cc4 + 2) * KPAD + kk] = (__bf16)wv.z;
            WldsT[(cc4 + 3) * KPAD + kk] = (__bf16)wv.w;
        }
        __syncthreads();

        for (int k0 = 0; k0 < kslab; k0 += 32) {
            // A fragment 16x32 bf16 (contiguous K, matching B):
            //   lane<16 : K {k0..k0+15}
            //   lane>=16: K {k0+16..k0+31}
            const int kb0 = ks + k0 + half * 16;
            const int kb1 = kb0 + 8;
            const float4 f0 = *(const float4*)(Arow + kb0);
            const float4 f1 = *(const float4*)(Arow + kb0 + 4);
            const float4 f2 = *(const float4*)(Arow + kb1);
            const float4 f3 = *(const float4*)(Arow + kb1 + 4);
            v16bf a;
            a[0]  = (__bf16)f0.x; a[1]  = (__bf16)f0.y;
            a[2]  = (__bf16)f0.z; a[3]  = (__bf16)f0.w;
            a[4]  = (__bf16)f1.x; a[5]  = (__bf16)f1.y;
            a[6]  = (__bf16)f1.z; a[7]  = (__bf16)f1.w;
            a[8]  = (__bf16)f2.x; a[9]  = (__bf16)f2.y;
            a[10] = (__bf16)f2.z; a[11] = (__bf16)f2.w;
            a[12] = (__bf16)f3.x; a[13] = (__bf16)f3.y;
            a[14] = (__bf16)f3.z; a[15] = (__bf16)f3.w;

#pragma unroll
            for (int t = 0; t < 8; ++t) {
                // B fragment 32x16: lane column c = t*16+nIn, 16 contiguous K
                // starting at k0 + half*16  -> two ds_load_b128.
                const int cIdx = t * 16 + nIn;
                const v8bf* bp =
                    (const v8bf*)&WldsT[cIdx * KPAD + k0 + half * 16];
                const v8bf blo = bp[0];
                const v8bf bhi = bp[1];
                v16bf b;
#pragma unroll
                for (int i = 0; i < 8; ++i) { b[i] = blo[i]; b[8 + i] = bhi[i]; }

                acc[t] = __builtin_amdgcn_wmma_f32_16x16x32_bf16(
                    false, a, false, b, (short)0, acc[t], false, false);
            }
        }
        __syncthreads();
    }

    // Store D: VGPR r -> row = rowBase + half*8 + r, col = colBase + t*16 + nIn
#pragma unroll
    for (int t = 0; t < 8; ++t) {
        const int n = colBase + t * 16 + nIn;
        float add = 0.0f;
        if (bias) add += bias[n];
        if (pos)  add += pos[n];
#pragma unroll
        for (int i = 0; i < 8; ++i) {
            const int m = rowBase + half * 8 + i;
            if (m < M)
                out[(size_t)m * N + n] = acc[t][i] + add;
        }
    }
}

// ---------------------------------------------------------------------------
// Per-(edge, head) attention logit: dot(q[row,h,:], k[col,h,:]), clip, exp,
// scatter-add into softmax denominator. Node arrays (25.6 MB) live in L2.
// ---------------------------------------------------------------------------
__global__ __launch_bounds__(256)
void edge_att_kernel(const long long* __restrict__ rows,
                     const long long* __restrict__ cols,
                     const float* __restrict__ Q, const float* __restrict__ Kf,
                     float* __restrict__ expAtt, float* __restrict__ attNorm,
                     int E) {
    int t = blockIdx.x * blockDim.x + threadIdx.x;
    if (t >= E * HEADS) return;
    const int h = t & (HEADS - 1);
    const int e = t >> 3;
    const int r = (int)rows[e];
    const int c = (int)cols[e];
    const float* q = Q  + (size_t)r * D_MODEL + h * HDIM;
    const float* k = Kf + (size_t)c * D_MODEL + h * HDIM;
    float s = 0.0f;
#pragma unroll
    for (int i = 0; i < HDIM; ++i) s = fmaf(q[i], k[i], s);
    s = fminf(10.0f, fmaxf(-10.0f, s));
    float ex = __expf(s);
    expAtt[(size_t)e * HEADS + h] = ex;
    atomicAdd(&attNorm[(size_t)r * HEADS + h], ex);
}

// ---------------------------------------------------------------------------
// Per-(edge, head) weighted scatter of V into agg (segment_sum over rows).
// ---------------------------------------------------------------------------
__global__ __launch_bounds__(256)
void edge_agg_kernel(const long long* __restrict__ rows,
                     const long long* __restrict__ cols,
                     const float* __restrict__ V,
                     const float* __restrict__ expAtt,
                     const float* __restrict__ attNorm,
                     float* __restrict__ agg, int E) {
    int t = blockIdx.x * blockDim.x + threadIdx.x;
    if (t >= E * HEADS) return;
    const int h = t & (HEADS - 1);
    const int e = t >> 3;
    const int r = (int)rows[e];
    const int c = (int)cols[e];
    const float w = expAtt[(size_t)e * HEADS + h] /
                    (attNorm[(size_t)r * HEADS + h] + 1e-8f);
    const float* v = V + (size_t)c * D_MODEL + h * HDIM;
    float* o = agg + (size_t)r * D_MODEL + h * HDIM;
#pragma unroll
    for (int i = 0; i < HDIM; ++i) atomicAdd(&o[i], w * v[i]);
}

// ---------------------------------------------------------------------------
// Residual add + LayerNorm. One wave32 per node; 4 elements per lane;
// wave reductions via shfl_xor.
// ---------------------------------------------------------------------------
__global__ __launch_bounds__(256)
void ln_kernel(const float* __restrict__ agg, const float* __restrict__ resid,
               const float* __restrict__ s, const float* __restrict__ b,
               float* __restrict__ out, int N) {
    const int gwave = (blockIdx.x * blockDim.x + threadIdx.x) >> 5;
    const int lane  = threadIdx.x & 31;
    if (gwave >= N) return;
    const float* a = agg   + (size_t)gwave * D_MODEL;
    const float* r = resid + (size_t)gwave * D_MODEL;

    float x[4];
    float sum = 0.0f;
#pragma unroll
    for (int j = 0; j < 4; ++j) {
        x[j] = a[lane + 32 * j] + r[lane + 32 * j];
        sum += x[j];
    }
#pragma unroll
    for (int off = 16; off > 0; off >>= 1) sum += __shfl_xor(sum, off, 32);
    const float mu = sum * (1.0f / D_MODEL);

    float var = 0.0f;
#pragma unroll
    for (int j = 0; j < 4; ++j) {
        float d = x[j] - mu;
        var += d * d;
    }
#pragma unroll
    for (int off = 16; off > 0; off >>= 1) var += __shfl_xor(var, off, 32);
    const float inv = rsqrtf(var * (1.0f / D_MODEL) + 1e-6f);

#pragma unroll
    for (int j = 0; j < 4; ++j) {
        const int c = lane + 32 * j;
        out[(size_t)gwave * D_MODEL + c] = (x[j] - mu) * inv * s[c] + b[c];
    }
}

__global__ __launch_bounds__(256)
void zero_kernel(float* __restrict__ p, long long n) {
    long long i = (long long)blockIdx.x * blockDim.x + threadIdx.x;
    if (i < n) p[i] = 0.0f;
}

// ---------------------------------------------------------------------------
extern "C" void kernel_launch(void* const* d_in, const int* in_sizes, int n_in,
                              void* d_out, int out_size, void* d_ws, size_t ws_size,
                              hipStream_t stream) {
    const float*     x     = (const float*)d_in[0];
    const long long* ei    = (const long long*)d_in[1];
    const float*     W_P_w = (const float*)d_in[2];
    const float*     W_P_b = (const float*)d_in[3];
    const float*     W_pos = (const float*)d_in[4];
    const float* qw[2] = {(const float*)d_in[5],  (const float*)d_in[10]};
    const float* kw[2] = {(const float*)d_in[6],  (const float*)d_in[11]};
    const float* vw[2] = {(const float*)d_in[7],  (const float*)d_in[12]};
    const float* ls[2] = {(const float*)d_in[8],  (const float*)d_in[13]};
    const float* lb[2] = {(const float*)d_in[9],  (const float*)d_in[14]};
    const float*     inv_w = (const float*)d_in[15];
    const float*     inv_b = (const float*)d_in[16];

    const int N    = in_sizes[0] / 256;   // 50000 nodes
    const int E    = in_sizes[1] / 2;     // 800000 edges
    const int GOUT = in_sizes[16];        // 256

    const long long* rows = ei;
    const long long* cols = ei + E;

    // Workspace carve-up (floats). ~155 MB needed.
    float* ws = (float*)d_ws;
    size_t off = 0;
    float* buf0 = ws + off; off += (size_t)N * D_MODEL;   // embeds / ping
    float* buf1 = ws + off; off += (size_t)N * D_MODEL;   // pong
    float* Qb   = ws + off; off += (size_t)N * D_MODEL;   // Q (reused as agg)
    float* Kb   = ws + off; off += (size_t)N * D_MODEL;
    float* Vb   = ws + off; off += (size_t)N * D_MODEL;
    float* attN = ws + off; off += (size_t)N * HEADS;
    float* expA = ws + off; off += (size_t)E * HEADS;
    float* agg  = Qb;  // reuse after attention logits are done

    const dim3 blk(256);
    const dim3 gNode128((N + 127) / 128, 1);
    const dim3 gNodeOut((N + 127) / 128, (GOUT + 127) / 128);
    const int  eBlocks  = (E * HEADS + 255) / 256;
    const int  nBlocks8 = ((long long)N * HEADS + 255) / 256;
    const int  nBlocksD = ((long long)N * D_MODEL + 255) / 256;
    const int  lnBlocks = ((long long)N * 32 + 255) / 256;

    // 1) Input projection + bias + positional embedding
    gemm_wmma_bf16<<<gNode128, blk, 0, stream>>>(x, W_P_w, W_P_b, W_pos, buf0,
                                                 N, 256, D_MODEL);

    float* lin[2]  = {buf0, buf1};
    float* lout[2] = {buf1, buf0};
    for (int L = 0; L < 2; ++L) {
        const float* in = lin[L];
        float* outp = lout[L];
        // Q/K/V node-level GEMMs
        gemm_wmma_bf16<<<gNode128, blk, 0, stream>>>(in, qw[L], nullptr, nullptr,
                                                     Qb, N, D_MODEL, D_MODEL);
        gemm_wmma_bf16<<<gNode128, blk, 0, stream>>>(in, kw[L], nullptr, nullptr,
                                                     Kb, N, D_MODEL, D_MODEL);
        gemm_wmma_bf16<<<gNode128, blk, 0, stream>>>(in, vw[L], nullptr, nullptr,
                                                     Vb, N, D_MODEL, D_MODEL);
        // scatter-softmax denominator
        zero_kernel<<<nBlocks8, blk, 0, stream>>>(attN, (long long)N * HEADS);
        edge_att_kernel<<<eBlocks, blk, 0, stream>>>(rows, cols, Qb, Kb,
                                                     expA, attN, E);
        // aggregate (agg aliases Qb -> zero only after edge_att consumed Q)
        zero_kernel<<<nBlocksD, blk, 0, stream>>>(agg, (long long)N * D_MODEL);
        edge_agg_kernel<<<eBlocks, blk, 0, stream>>>(rows, cols, Vb, expA, attN,
                                                     agg, E);
        // residual + LayerNorm
        ln_kernel<<<lnBlocks, blk, 0, stream>>>(agg, in, ls[L], lb[L], outp, N);
    }

    // 2) Output projection (layer-2 output lives in buf0)
    gemm_wmma_bf16<<<gNodeOut, blk, 0, stream>>>(buf0, inv_w, inv_b, nullptr,
                                                 (float*)d_out, N, D_MODEL, GOUT);
}
